// CausalDecayMemory_71347996721401
// MI455X (gfx1250) — matrix operand order repaired and checked by reading, not compiled, not run on hardware
//
#include <hip/hip_runtime.h>
#include <hip/hip_bf16.h>
#include <math.h>

// ---------------------------------------------------------------------------
// CDNA5 (gfx1250) wave32 WMMA implementation of CausalDecayMemory.
//   out = ((q k^T * scale * decay_weights) v) Wo^T * out_scale
// All GEMMs run through v_wmma_f32_16x16x32_bf16; the TxT score matrix is
// tiled and kept in LDS (never hits HBM).  LDS staging uses the CDNA5 async
// copy unit (global_load_async_to_lds_b128) with DOUBLE BUFFERING: the next
// K-tile is in flight while WMMAs consume the current one, synchronized with
// partial s_wait_asynccnt (in-order completion of async loads).
// ---------------------------------------------------------------------------

typedef __attribute__((ext_vector_type(16))) __bf16 v16bf;
typedef __attribute__((ext_vector_type(8)))  __bf16 v8bf;
typedef __attribute__((ext_vector_type(8)))  float  v8f;
typedef unsigned short u16;

#define DEV static __device__ __forceinline__

DEV u16 f32_to_bf16_bits(float f) {
  unsigned int u = __float_as_uint(f);
  u += 0x7FFFu + ((u >> 16) & 1u);          // round-to-nearest-even
  return (u16)(u >> 16);
}
DEV __bf16 bits_to_bf16(u16 s) {
  union { u16 u; __bf16 b; } x; x.u = s; return x.b;
}
DEV v8f zero8() {
  v8f z;
#pragma unroll
  for (int i = 0; i < 8; ++i) z[i] = 0.0f;
  return z;
}

// --- CDNA5 async copy: 16B global -> LDS, tracked by ASYNCcnt --------------
DEV void async_ld_b128(void* lds_ptr, const void* gaddr) {
  unsigned lds_addr = (unsigned)(uintptr_t)lds_ptr;
  unsigned long long ga = (unsigned long long)(uintptr_t)gaddr;
  asm volatile("global_load_async_to_lds_b128 %0, %1, off"
               :: "v"(lds_addr), "v"(ga)
               : "memory");
}
DEV void wait_async0() { asm volatile("s_wait_asynccnt 0" ::: "memory"); }
// wait until at most 3 async ops outstanding (== the 3 chunks of the tile
// being prefetched; the current tile's 3 chunks are then complete, since
// async loads return done in order).
DEV void wait_async3() { asm volatile("s_wait_asynccnt 3" ::: "memory"); }

// A-fragment (16x32 bf16): lane half 0 holds K {k0..k0+15},
// lane half 1 holds K {k0+16..k0+31}; row m = lane%16.
DEV v16bf load_a_frag(const u16* row, int k0, int half) {
  const v8bf lo = *(const v8bf*)(row + k0 + 16 * half);
  const v8bf hi = *(const v8bf*)(row + k0 + 16 * half + 8);
  v16bf f;
#pragma unroll
  for (int i = 0; i < 8; ++i) { f[i] = lo[i]; f[8 + i] = hi[i]; }
  return f;
}
// B-fragment (32x16 bf16): lane owns column n = lane%16; lanes 0-15 hold
// K 0..15, lanes 16-31 hold K 16..31 -> one contiguous 16-element run.
DEV v16bf load_b_frag(const u16* p) {
  const v8bf lo = *(const v8bf*)(p);
  const v8bf hi = *(const v8bf*)(p + 8);
  v16bf f;
#pragma unroll
  for (int i = 0; i < 8; ++i) { f[i] = lo[i]; f[8 + i] = hi[i]; }
  return f;
}
DEV v8f wmma_bf16(v16bf a, v16bf b, v8f c) {
  return __builtin_amdgcn_wmma_f32_16x16x32_bf16(
      /*neg_a=*/false, a, /*neg_b=*/false, b,
      /*c_mod=*/(short)0, c, /*reuse_a=*/false, /*reuse_b=*/false);
}

// ---------------------------------------------------------------------------
// f32 -> bf16 conversion (grid-stride)
// ---------------------------------------------------------------------------
__global__ void cvt_f32_bf16(const float* __restrict__ in,
                             u16* __restrict__ out, size_t n) {
  size_t i = (size_t)blockIdx.x * blockDim.x + threadIdx.x;
  size_t stride = (size_t)gridDim.x * blockDim.x;
  for (; i < n; i += stride) out[i] = f32_to_bf16_bits(in[i]);
}

// ---------------------------------------------------------------------------
// C = A(MxK) * Bw(NxK)^T  with bf16 inputs.  Workgroup tile 64(M) x 128(N),
// 8 waves in a 2x4 grid, each wave owns a 32x32 patch (2x2 WMMA tiles).
// Double-buffered async LDS staging (3 x b128 chunks per thread per tile).
// mode 0: C bf16 row-major [M,N]
// mode 1: C bf16 transposed per batch: [b][n][t], b = gm >> tLog2 (T = 2^tLog2)
// mode 2: C f32 row-major scaled by *alpha_p
// ---------------------------------------------------------------------------
__global__ __launch_bounds__(256)
void gemm_bf16_nt(const u16* __restrict__ A, const u16* __restrict__ Bw,
                  void* __restrict__ Cout, int M, int N, int K,
                  int mode, const float* __restrict__ alpha_p, int tLog2) {
  __shared__ __align__(16) u16 As[2][64 * 32];    //  2 x 4 KB
  __shared__ __align__(16) u16 Bs[2][128 * 32];   //  2 x 8 KB

  const int tid  = threadIdx.x;
  const int lane = tid & 31;
  const int w    = tid >> 5;
  const int wm   = w >> 2;              // 0..1
  const int wn   = w & 3;               // 0..3
  const int half = lane >> 4;
  const int l16  = lane & 15;
  const int m0   = blockIdx.y * 64;
  const int n0   = blockIdx.x * 128;
  const float alpha = alpha_p ? *alpha_p : 1.0f;

  v8f acc[2][2];
#pragma unroll
  for (int i = 0; i < 2; ++i)
#pragma unroll
    for (int j = 0; j < 2; ++j) acc[i][j] = zero8();

  // per-thread staging addresses: chunk t of As is As[buf][t*8] (rows of 32).
  const int arow = tid >> 2, achk = tid & 3;
  const u16* agp = A  + (size_t)(m0 + arow) * K + achk * 8;
  const int brow0 = tid >> 2,          bchk0 = tid & 3;          // idx = tid
  const int brow1 = (tid + 256) >> 2,  bchk1 = tid & 3;          // idx = tid+256
  const u16* bgp0 = Bw + (size_t)(n0 + brow0) * K + bchk0 * 8;
  const u16* bgp1 = Bw + (size_t)(n0 + brow1) * K + bchk1 * 8;

#define STAGE(buf, kb)                                                   \
  do {                                                                   \
    async_ld_b128(&As[buf][(size_t)tid * 8],         agp  + (kb));       \
    async_ld_b128(&Bs[buf][(size_t)tid * 8],         bgp0 + (kb));       \
    async_ld_b128(&Bs[buf][(size_t)(tid + 256) * 8], bgp1 + (kb));       \
  } while (0)

  STAGE(0, 0);
  int pb = 0;
  for (int kb = 0; kb < K; kb += 32, pb ^= 1) {
    const bool more = (kb + 32 < K);
    if (more) STAGE(pb ^ 1, kb + 32);   // next tile in flight behind WMMAs
    if (more) wait_async3(); else wait_async0();
    __syncthreads();

#pragma unroll
    for (int mi = 0; mi < 2; ++mi) {
      const u16* ar = &As[pb][(wm * 32 + mi * 16 + l16) * 32];
      v16bf af = load_a_frag(ar, 0, half);
#pragma unroll
      for (int ni = 0; ni < 2; ++ni) {
        const u16* br = &Bs[pb][(wn * 32 + ni * 16 + l16) * 32 + 16 * half];
        acc[mi][ni] = wmma_bf16(af, load_b_frag(br), acc[mi][ni]);
      }
    }
    __syncthreads();
  }
#undef STAGE

  // epilogue: C/D layout -> M = r + 8*half, N = l16
#pragma unroll
  for (int mi = 0; mi < 2; ++mi)
#pragma unroll
    for (int ni = 0; ni < 2; ++ni)
#pragma unroll
      for (int r = 0; r < 8; ++r) {
        int gm = m0 + wm * 32 + mi * 16 + r + 8 * half;
        int gn = n0 + wn * 32 + ni * 16 + l16;
        float v = acc[mi][ni][r];
        if (mode == 0) {
          ((u16*)Cout)[(size_t)gm * N + gn] = f32_to_bf16_bits(v);
        } else if (mode == 1) {
          int bb = gm >> tLog2, rr = gm & ((1 << tLog2) - 1);
          ((u16*)Cout)[((size_t)bb * N + gn) << tLog2 | rr] = f32_to_bf16_bits(v);
        } else {
          ((float*)Cout)[(size_t)gm * N + gn] = v * alpha;
        }
      }
}

// ---------------------------------------------------------------------------
// Fused decayed attention: for one (batch, 16-row q tile), stream over 32-wide
// s-tiles.  8 waves split the K=1024 score reduction (ds_add_f32 into an LDS
// 16x32 f32 tile), then each wave applies decay weights and accumulates its
// private 128-column slice of retrieved.  Scores never leave HBM/LDS.
// ---------------------------------------------------------------------------
__global__ __launch_bounds__(256)
void attn_decay(const u16* __restrict__ qb, const u16* __restrict__ kb,
                const u16* __restrict__ vT, u16* __restrict__ rb,
                const float* __restrict__ decay_logit_p,
                int Bn, int T, int d) {
  __shared__ __align__(16) u16  qs[16 * 1024];  // 32 KB: q tile (16 x d) bf16
  __shared__ __align__(16) float sc[16 * 32];   //  2 KB: f32 score tile

  const int tid  = threadIdx.x;
  const int lane = tid & 31;
  const int w    = tid >> 5;
  const int half = lane >> 4;
  const int l16  = lane & 15;

  const int tilesPerB = T >> 4;
  const int b  = blockIdx.x / tilesPerB;
  const int t0 = (blockIdx.x % tilesPerB) << 4;

  const float dl    = *decay_logit_p;
  const float decay = 1.0f / (1.0f + expf(-dl));
  const float l2d   = log2f(decay);
  const float scale = rsqrtf((float)d);

  { // async stage of the q tile into LDS (16B chunks via async copy unit)
    const u16* src = qb + (size_t)(b * T + t0) * d;
    const int nchunks = 16 * d / 8;             // 2048
    for (int i = tid; i < nchunks; i += 256)
      async_ld_b128(qs + (size_t)i * 8, src + (size_t)i * 8);
    wait_async0();
  }
  __syncthreads();

  const int d0  = w * 128;                       // retrieved column slice
  const int kw0 = w * 128;                       // score K-chunk of this wave
  v8f racc[8];
#pragma unroll
  for (int i = 0; i < 8; ++i) racc[i] = zero8();

  const u16* kbat = kb + (size_t)b * T * d;      // [T][d]
  const u16* vbat = vT + (size_t)b * d * T;      // [d][T]
  const int i_row = t0 + l16;

  for (int s0 = t0 & ~31; s0 < T; s0 += 32) {
    // hint the next s-tile's k rows toward the caches (global_prefetch_b8)
    if (s0 + 32 < T)
      __builtin_prefetch(kbat + (size_t)(s0 + 32 + lane) * d + kw0, 0, 1);

    // (1) zero the LDS score tile
    for (int i = tid; i < 16 * 32; i += 256) sc[i] = 0.0f;
    __syncthreads();

    // (2) partial scores over this wave's 128-wide K chunk
    const u16* qrow = qs + l16 * d;
#pragma unroll
    for (int nt = 0; nt < 2; ++nt) {
      v8f p = zero8();
#pragma unroll
      for (int ks = 0; ks < 4; ++ks) {
        int k0 = kw0 + 32 * ks;
        v16bf af = load_a_frag(qrow, k0, half);
        int j = s0 + nt * 16 + l16;              // B column n = lane%16
        v16bf bfg = load_b_frag(kbat + (size_t)j * d + k0 + 16 * half);
        p = wmma_bf16(af, bfg, p);
      }
#pragma unroll
      for (int r = 0; r < 8; ++r)
        atomicAdd(&sc[(r + 8 * half) * 32 + nt * 16 + l16], p[r]);
    }
    __syncthreads();

    // (3) apply decay weights, repack a bf16 A-fragment of the score tile
    v16bf wa;
#pragma unroll
    for (int e = 0; e < 16; ++e) {
      int kk = 16 * half + e;
      int j = s0 + kk;
      float val = sc[l16 * 32 + kk];
      float wgt = (j > i_row)
                      ? exp2f(l2d * (float)(j - i_row - 1)) * scale
                      : 0.0f;
      wa[e] = bits_to_bf16(f32_to_bf16_bits(val * wgt));
    }

    // (4) retrieved += weighted_scores(16x32) @ v(32 x 128-slice)
#pragma unroll
    for (int nt = 0; nt < 8; ++nt) {
      int n = d0 + nt * 16 + l16;
      v16bf vb = load_b_frag(vbat + (size_t)n * T + s0 + 16 * half);
      racc[nt] = wmma_bf16(wa, vb, racc[nt]);
    }
    __syncthreads();   // all reads of sc done before next-iter zeroing
  }

  // store retrieved (bf16 row-major [B*T, d])
#pragma unroll
  for (int nt = 0; nt < 8; ++nt)
#pragma unroll
    for (int r = 0; r < 8; ++r) {
      int gm = t0 + r + 8 * half;
      int gn = d0 + nt * 16 + l16;
      rb[(size_t)(b * T + gm) * d + gn] = f32_to_bf16_bits(racc[nt][r]);
    }
}

// ---------------------------------------------------------------------------
extern "C" void kernel_launch(void* const* d_in, const int* in_sizes, int n_in,
                              void* d_out, int out_size, void* d_ws,
                              size_t ws_size, hipStream_t stream) {
  (void)in_sizes; (void)n_in; (void)out_size; (void)ws_size;

  const float* x  = (const float*)d_in[0];
  const float* Wq = (const float*)d_in[1];
  const float* Wk = (const float*)d_in[2];
  const float* Wv = (const float*)d_in[3];
  const float* Wo = (const float*)d_in[4];
  const float* dl = (const float*)d_in[5];   // decay_logit (device scalar)
  const float* os = (const float*)d_in[6];   // out_scale   (device scalar)
  float* out = (float*)d_out;

  const int Bn = 4, T = 4096, d = 1024;
  const int tLog2 = 12;                      // log2(T)
  const int M = Bn * T;                      // 16384
  const size_t MD = (size_t)M * d;           // 16.7M elems
  const size_t DD = (size_t)d * d;

  char* ws = (char*)d_ws;
  size_t off = 0;
  u16* xb  = (u16*)(ws + off); off += MD * sizeof(u16);   // x  bf16
  u16* qb  = (u16*)(ws + off); off += MD * sizeof(u16);   // q  bf16 [M,d]
  u16* kbf = (u16*)(ws + off); off += MD * sizeof(u16);   // k  bf16 [M,d]
  u16* vTb = (u16*)(ws + off); off += MD * sizeof(u16);   // v^T bf16 [B][d][T]
  u16* rb  = (u16*)(ws + off); off += MD * sizeof(u16);   // retrieved bf16
  u16* wqb = (u16*)(ws + off); off += DD * sizeof(u16);
  u16* wkb = (u16*)(ws + off); off += DD * sizeof(u16);
  u16* wvb = (u16*)(ws + off); off += DD * sizeof(u16);
  u16* wob = (u16*)(ws + off); off += DD * sizeof(u16);

  // 1) precision down-convert
  cvt_f32_bf16<<<2048, 256, 0, stream>>>(x,  xb,  MD);
  cvt_f32_bf16<<<256,  256, 0, stream>>>(Wq, wqb, DD);
  cvt_f32_bf16<<<256,  256, 0, stream>>>(Wk, wkb, DD);
  cvt_f32_bf16<<<256,  256, 0, stream>>>(Wv, wvb, DD);
  cvt_f32_bf16<<<256,  256, 0, stream>>>(Wo, wob, DD);

  // 2) projections (q,k row-major; v stored transposed per batch)
  dim3 gg(d / 128, M / 64);                  // (8, 256)
  gemm_bf16_nt<<<gg, 256, 0, stream>>>(xb, wqb, qb,  M, d, d, 0, nullptr, tLog2);
  gemm_bf16_nt<<<gg, 256, 0, stream>>>(xb, wkb, kbf, M, d, d, 0, nullptr, tLog2);
  gemm_bf16_nt<<<gg, 256, 0, stream>>>(xb, wvb, vTb, M, d, d, 1, nullptr, tLog2);

  // 3) fused decayed attention (scores live only in LDS)
  attn_decay<<<Bn * (T / 16), 256, 0, stream>>>(qb, kbf, vTb, rb, dl, Bn, T, d);

  // 4) output projection, f32 * out_scale
  gemm_bf16_nt<<<gg, 256, 0, stream>>>(rb, wob, out, M, d, d, 2, os, tLog2);
}
